// ClusterFeatureNetWithAttention_69303592288676
// MI455X (gfx1250) — matrix operand order repaired and checked by reading, not compiled, not run on hardware
//
#include <hip/hip_runtime.h>
#include <hip/hip_bf16.h>

typedef _Float16 v8h  __attribute__((ext_vector_type(8)));
typedef _Float16 v16h __attribute__((ext_vector_type(16)));
typedef float    v8f  __attribute__((ext_vector_type(8)));

#define C_CL 512
#define P_PT 256
#define D_IN 256
#define H_HID 512
#define O_OUT 256
#define XLD 264   // padded row stride (halves) for LDS matrices: 264*2B=528B -> 4-bank shift per row

// ---------------- fragment loaders (CDNA5 WMMA 16x16x32 f16 layouts) ----------------
// A (16x32, MxK): lane L -> row = L&15; K chunks: [kbase..kbase+7] and [kbase+16..kbase+23], kbase=(L>>4)*8
__device__ __forceinline__ v16h ld_a_frag(const _Float16* base, int ld, int r0, int k0,
                                          int lrow, int lhalf) {
  const _Float16* p = base + (r0 + lrow) * ld + k0 + lhalf * 8;
  v8h a0 = *(const v8h*)p;
  v8h a1 = *(const v8h*)(p + 16);
  v16h a;
#pragma unroll
  for (int i = 0; i < 8; ++i) { a[i] = a0[i]; a[8 + i] = a1[i]; }
  return a;
}

// B (32x16, KxN) loaded from Bt where Bt[n][k] = B[k][n]:
// lane L -> col N = L&15; K chunks: [kbase..kbase+7] and [kbase+16..kbase+23], kbase=(L>>4)*8
__device__ __forceinline__ v16h ld_bt_frag(const _Float16* bt, int ld, int n0, int k0,
                                           int lrow, int lhalf) {
  const _Float16* p = bt + (n0 + lrow) * ld + k0 + lhalf * 8;
  v8h b0 = *(const v8h*)p;
  v8h b1 = *(const v8h*)(p + 16);
  v16h b;
#pragma unroll
  for (int i = 0; i < 8; ++i) { b[i] = b0[i]; b[8 + i] = b1[i]; }
  return b;
}

// ---------------- weight preprocessing kernels ----------------
// Mt[n][k] = (Wq Wk^T)[k][n] / sqrt(H)   (f16, stored transposed for B-operand loads)
__global__ void prep_mt(const float* __restrict__ Wq, const float* __restrict__ Wk,
                        _Float16* __restrict__ Mt) {
  int n = blockIdx.x, k = threadIdx.x;
  const float* wq = Wq + k * H_HID;
  const float* wk = Wk + n * H_HID;
  float s = 0.f;
  for (int h = 0; h < H_HID; ++h) s += wq[h] * wk[h];
  Mt[n * D_IN + k] = (_Float16)(s * 0.04419417382415922f);  // 1/sqrt(512)
}

// kb[k] = (Wk bq)[k] / sqrt(H)  (column softmax bias; zero when bq==0)
__global__ void prep_kb(const float* __restrict__ Wk, const float* __restrict__ bq,
                        float* __restrict__ kb) {
  int k = threadIdx.x;
  const float* wk = Wk + k * H_HID;
  float s = 0.f;
  for (int h = 0; h < H_HID; ++h) s += wk[h] * bq[h];
  kb[k] = s * 0.04419417382415922f;
}

// Wvft[o][d] = (Wv Wf)[d][o]  (fused V->Wf projection, transposed for dot loops)
__global__ void prep_wvft(const float* __restrict__ Wv, const float* __restrict__ Wf,
                          float* __restrict__ Wvft) {
  int o = blockIdx.x, d = threadIdx.x;
  const float* wv = Wv + d * H_HID;
  float s = 0.f;
  for (int h = 0; h < H_HID; ++h) s += wv[h] * Wf[h * O_OUT + o];
  Wvft[o * D_IN + d] = s;
}

// c1[o] = (bv Wf)[o] + bf[o]
__global__ void prep_c1(const float* __restrict__ bv, const float* __restrict__ Wf,
                        const float* __restrict__ bf, float* __restrict__ c1) {
  int o = threadIdx.x;
  float s = 0.f;
  for (int h = 0; h < H_HID; ++h) s += bv[h] * Wf[h * O_OUT + o];
  c1[o] = s + bf[o];
}

__global__ void prep_w1t(const float* __restrict__ W1, float* __restrict__ W1t) {
  int h = blockIdx.x, o = threadIdx.x;
  W1t[h * O_OUT + o] = W1[o * H_HID + h];
}
__global__ void prep_w2t(const float* __restrict__ W2, float* __restrict__ W2t) {
  int o = blockIdx.x, h = threadIdx.x;
  W2t[o * H_HID + h] = W2[h * O_OUT + o];
}

// ---------------- main per-cluster fused kernel ----------------
__global__ __launch_bounds__(256) void cluster_attn_kernel(
    const float* __restrict__ src_feats, const float* __restrict__ tgt_feats,
    const float* __restrict__ src_pts, const float* __restrict__ tgt_pts,
    const int* __restrict__ idx_src, const int* __restrict__ idx_tgt,
    const _Float16* __restrict__ Mt, const float* __restrict__ kb,
    const float* __restrict__ Wvft, const float* __restrict__ c1,
    const float* __restrict__ W1t, const float* __restrict__ W2t,
    const float* __restrict__ b1, const float* __restrict__ b2,
    float* __restrict__ out) {
  extern __shared__ char smem[];
  _Float16* xs = (_Float16*)smem;          // [256][XLD] gathered features, f16
  _Float16* Ts = xs + 256 * XLD;           // [256][XLD] T = x @ M, f16
  float* wsum = (float*)(Ts + 256 * XLD);  // [256] column means of attention (unnormalized)
  float* rs   = wsum + 256;                // [256] column softmax bias
  float* fps  = rs + 256;                  // [256] pooled feature (pre-proj)
  float* gs   = fps + 256;                 // [256] after Wv*Wf
  float* hs   = gs + 256;                  // [512] MLP hidden
  float* csum = hs + 512;                  // [4] coord sums

  const int tid = threadIdx.x;
  const int wave = tid >> 5, lane = tid & 31;
  const int lrow = lane & 15, lhalf = lane >> 4;

  const int b = blockIdx.x;
  const int side = b >> 9;   // 0=src, 1=tgt  (C=512)
  const int c = b & 511;

  const float* feats = side ? tgt_feats : src_feats;
  const float* pts   = side ? tgt_pts : src_pts;
  const int* idx     = (side ? idx_tgt : idx_src) + c * P_PT;

  wsum[tid] = 0.f;
  if (tid < 4) csum[tid] = 0.f;
  __syncthreads();

  __builtin_prefetch(Mt + tid * 256, 0, 3);  // warm Mt in cache (global_prefetch_b8)

  // ---- gather x rows (f32 -> f16 into LDS); one wave per row strip
  for (int p = wave; p < P_PT; p += 8) {
    long gp = idx[p];
    const float4* srcp = (const float4*)(feats + gp * (long)D_IN);
    float4 f0 = srcp[lane * 2];
    float4 f1 = srcp[lane * 2 + 1];
    v8h hv;
    hv[0] = (_Float16)f0.x; hv[1] = (_Float16)f0.y;
    hv[2] = (_Float16)f0.z; hv[3] = (_Float16)f0.w;
    hv[4] = (_Float16)f1.x; hv[5] = (_Float16)f1.y;
    hv[6] = (_Float16)f1.z; hv[7] = (_Float16)f1.w;
    *(v8h*)(xs + p * XLD + lane * 8) = hv;
  }

  // ---- coord mean reduction
  {
    long gp = idx[tid];
    float cx = pts[gp * 3 + 0], cy = pts[gp * 3 + 1], cz = pts[gp * 3 + 2];
#pragma unroll
    for (int m = 16; m >= 1; m >>= 1) {
      cx += __shfl_xor(cx, m, 32);
      cy += __shfl_xor(cy, m, 32);
      cz += __shfl_xor(cz, m, 32);
    }
    if (lane == 0) {
      atomicAdd(&csum[0], cx);
      atomicAdd(&csum[1], cy);
      atomicAdd(&csum[2], cz);
    }
  }
  __syncthreads();

  // ---- r[q] = x[q,:] . kb  (consumed after the T-stage barrier)
  {
    float s = 0.f;
    const _Float16* row = xs + tid * XLD;
    for (int d = 0; d < D_IN; ++d) s += (float)row[d] * kb[d];
    rs[tid] = s;
  }

  // ---- T = x @ M  via WMMA (each wave: two 16-row strips, 16 col tiles)
  for (int i2 = 0; i2 < 2; ++i2) {
    const int i = wave * 2 + i2;
    v8f acc[16] = {};
    for (int kk = 0; kk < 8; ++kk) {
      v16h a = ld_a_frag(xs, XLD, i * 16, kk * 32, lrow, lhalf);
#pragma unroll
      for (int j = 0; j < 16; ++j) {
        v16h bb = ld_bt_frag(Mt, D_IN, j * 16, kk * 32, lrow, lhalf);
        acc[j] = __builtin_amdgcn_wmma_f32_16x16x32_f16(false, a, false, bb,
                                                        (short)0, acc[j], false, false);
      }
    }
#pragma unroll
    for (int j = 0; j < 16; ++j)
#pragma unroll
      for (int r = 0; r < 8; ++r)
        Ts[(i * 16 + r + lhalf * 8) * XLD + j * 16 + lrow] = (_Float16)acc[j][r];
  }
  __syncthreads();

  // ---- S = T @ x^T strips + streaming softmax + attention column sums
  for (int i2 = 0; i2 < 2; ++i2) {
    const int i = wave * 2 + i2;
    v8f acc[16] = {};
    for (int kk = 0; kk < 8; ++kk) {
      v16h a = ld_a_frag(Ts, XLD, i * 16, kk * 32, lrow, lhalf);
#pragma unroll
      for (int j = 0; j < 16; ++j) {
        v16h bb = ld_bt_frag(xs, XLD, j * 16, kk * 32, lrow, lhalf);
        acc[j] = __builtin_amdgcn_wmma_f32_16x16x32_f16(false, a, false, bb,
                                                        (short)0, acc[j], false, false);
      }
    }
    // add column bias r[q]
#pragma unroll
    for (int j = 0; j < 16; ++j) {
      float rq = rs[j * 16 + lrow];
#pragma unroll
      for (int r = 0; r < 8; ++r) acc[j][r] += rq;
    }
    // per-row max (row lives in one 16-lane half across the 16 tiles)
    float mx[8], zi[8];
#pragma unroll
    for (int r = 0; r < 8; ++r) {
      float m = acc[0][r];
#pragma unroll
      for (int j = 1; j < 16; ++j) m = fmaxf(m, acc[j][r]);
      m = fmaxf(m, __shfl_xor(m, 1, 32));
      m = fmaxf(m, __shfl_xor(m, 2, 32));
      m = fmaxf(m, __shfl_xor(m, 4, 32));
      m = fmaxf(m, __shfl_xor(m, 8, 32));
      mx[r] = m;
    }
    // exp + per-row sum
#pragma unroll
    for (int r = 0; r < 8; ++r) {
      float z = 0.f;
#pragma unroll
      for (int j = 0; j < 16; ++j) {
        float e = __expf(acc[j][r] - mx[r]);
        acc[j][r] = e;
        z += e;
      }
      z += __shfl_xor(z, 1, 32);
      z += __shfl_xor(z, 2, 32);
      z += __shfl_xor(z, 4, 32);
      z += __shfl_xor(z, 8, 32);
      zi[r] = 1.0f / z;
    }
    // column partial sums of normalized attention -> wsum
#pragma unroll
    for (int j = 0; j < 16; ++j) {
      float cs = 0.f;
#pragma unroll
      for (int r = 0; r < 8; ++r) cs += acc[j][r] * zi[r];
      cs += __shfl_xor(cs, 16, 32);
      if (lhalf == 0) atomicAdd(&wsum[j * 16 + lrow], cs);
    }
  }
  __syncthreads();

  // ---- pooled feature: fp[d] = (1/P) * sum_q wsum[q] * x[q][d]
  {
    float s = 0.f;
    for (int q = 0; q < P_PT; ++q) s += wsum[q] * (float)xs[q * XLD + tid];
    fps[tid] = s * (1.0f / 256.0f);
  }
  __syncthreads();

  // ---- g = fp @ (Wv Wf) + (bv Wf + bf)
  {
    float s = c1[tid];
    const float* wv = Wvft + tid * D_IN;
    for (int d = 0; d < D_IN; ++d) s += fps[d] * wv[d];
    gs[tid] = s;
  }
  __syncthreads();

  // ---- hidden = relu(g @ W1 + b1)   (each thread: 2 of 512 outputs)
#pragma unroll
  for (int t = 0; t < 2; ++t) {
    int j = tid + t * 256;
    float s = b1[j];
    const float* w1 = W1t + j * O_OUT;
    for (int o = 0; o < O_OUT; ++o) s += gs[o] * w1[o];
    hs[j] = fmaxf(s, 0.f);
  }
  __syncthreads();

  // ---- feat = hidden @ W2 + b2
  {
    float s = b2[tid];
    const float* w2 = W2t + tid * H_HID;
    for (int h = 0; h < H_HID; ++h) s += hs[h] * w2[h];
    float* fo = out + (long)side * (C_CL * O_OUT) + c * O_OUT;
    fo[tid] = s;
  }
  if (tid < 3) {
    float* co = out + 2L * C_CL * O_OUT + side * (C_CL * 3) + c * 3;
    co[tid] = csum[tid] * (1.0f / 256.0f);
  }
}

extern "C" void kernel_launch(void* const* d_in, const int* in_sizes, int n_in,
                              void* d_out, int out_size, void* d_ws, size_t ws_size,
                              hipStream_t stream) {
  const float* src_feats = (const float*)d_in[0];
  const float* tgt_feats = (const float*)d_in[1];
  const float* src_pts   = (const float*)d_in[2];
  const float* tgt_pts   = (const float*)d_in[3];
  const int*   idx_src   = (const int*)d_in[4];
  const int*   idx_tgt   = (const int*)d_in[5];
  const float* Wq = (const float*)d_in[6];
  const float* bq = (const float*)d_in[7];
  const float* Wk = (const float*)d_in[8];
  // bk (d_in[9]) contributes only softmax-invariant terms; algebraically eliminated.
  const float* Wv = (const float*)d_in[10];
  const float* bv = (const float*)d_in[11];
  const float* Wf = (const float*)d_in[12];
  const float* bf = (const float*)d_in[13];
  const float* W1 = (const float*)d_in[14];
  const float* b1 = (const float*)d_in[15];
  const float* W2 = (const float*)d_in[16];
  const float* b2 = (const float*)d_in[17];
  float* out = (float*)d_out;

  // workspace layout
  char* w = (char*)d_ws;
  _Float16* Mt = (_Float16*)w;  w += (size_t)D_IN * D_IN * sizeof(_Float16);
  float* kb    = (float*)w;     w += D_IN * sizeof(float);
  float* Wvft  = (float*)w;     w += (size_t)O_OUT * D_IN * sizeof(float);
  float* c1    = (float*)w;     w += O_OUT * sizeof(float);
  float* W1t   = (float*)w;     w += (size_t)H_HID * O_OUT * sizeof(float);
  float* W2t   = (float*)w;     w += (size_t)O_OUT * H_HID * sizeof(float);

  hipLaunchKernelGGL(prep_mt,   dim3(D_IN),  dim3(D_IN),  0, stream, Wq, Wk, Mt);
  hipLaunchKernelGGL(prep_kb,   dim3(1),     dim3(D_IN),  0, stream, Wk, bq, kb);
  hipLaunchKernelGGL(prep_wvft, dim3(O_OUT), dim3(D_IN),  0, stream, Wv, Wf, Wvft);
  hipLaunchKernelGGL(prep_c1,   dim3(1),     dim3(O_OUT), 0, stream, bv, Wf, bf, c1);
  hipLaunchKernelGGL(prep_w1t,  dim3(H_HID), dim3(O_OUT), 0, stream, W1, W1t);
  hipLaunchKernelGGL(prep_w2t,  dim3(O_OUT), dim3(H_HID), 0, stream, W2, W2t);

  const size_t smem = (size_t)2 * 256 * XLD * sizeof(_Float16) +
                      (256 * 4 + 512 + 4) * sizeof(float);  // ~276 KB < 320 KB WGP LDS
  (void)hipFuncSetAttribute((const void*)cluster_attn_kernel,
                            hipFuncAttributeMaxDynamicSharedMemorySize, (int)smem);
  hipLaunchKernelGGL(cluster_attn_kernel, dim3(2 * C_CL), dim3(256), smem, stream,
                     src_feats, tgt_feats, src_pts, tgt_pts, idx_src, idx_tgt,
                     Mt, kb, Wvft, c1, W1t, W2t, b1, b2, out);
}
